// MultiHeadAttention_2070174237247
// MI455X (gfx1250) — matrix operand order repaired and checked by reading, not compiled, not run on hardware
//
#include <hip/hip_runtime.h>
#include <math.h>
#include <stdint.h>

#define BB 2
#define SS 2048
#define DD 1024
#define HH 16
#define HDm 64
#define MM (BB*SS)

typedef __attribute__((ext_vector_type(16))) __bf16 v16bf;
typedef __attribute__((ext_vector_type(8)))  float  v8f;

union Frag16 { v16bf v; uint4 u[2]; unsigned short s[16]; };

__device__ __forceinline__ unsigned short f2bf(float f) {
  unsigned int u = __float_as_uint(f);
  u += 0x7FFFu + ((u >> 16) & 1u);
  return (unsigned short)(u >> 16);
}

// async global->LDS copy of 16 bytes (per lane), tracked by ASYNCcnt
__device__ __forceinline__ void async_b128(unsigned ldsOff, const void* gptr) {
  unsigned long long g = (unsigned long long)(uintptr_t)gptr;
  asm volatile("global_load_async_to_lds_b128 %0, %1, off"
               :: "v"(ldsOff), "v"(g) : "memory");
}
__device__ __forceinline__ void wait_async0() {
  asm volatile("s_wait_asynccnt 0x0" ::: "memory");
}

// ---------------- f32 -> bf16 elementwise (vec4) ----------------
__global__ void cvt_f32_bf16(const float* __restrict__ in,
                             unsigned short* __restrict__ out, int n4) {
  int i = blockIdx.x * blockDim.x + threadIdx.x;
  if (i >= n4) return;
  float4 f = ((const float4*)in)[i];
  union { unsigned short s[4]; ushort4 u; } p;
  p.s[0] = f2bf(f.x); p.s[1] = f2bf(f.y); p.s[2] = f2bf(f.z); p.s[3] = f2bf(f.w);
  ((ushort4*)out)[i] = p.u;
}

// ---------------- W[k][n] f32 -> Wt[n][k] bf16 (LDS tile transpose) ----------
__global__ void transpose_w_bf16(const float* s0, const float* s1,
                                 const float* s2, const float* s3,
                                 unsigned short* d0, unsigned short* d1,
                                 unsigned short* d2, unsigned short* d3) {
  __shared__ float tile[32][33];
  const float* src = (blockIdx.z == 0) ? s0 : (blockIdx.z == 1) ? s1
                   : (blockIdx.z == 2) ? s2 : s3;
  unsigned short* dst = (blockIdx.z == 0) ? d0 : (blockIdx.z == 1) ? d1
                      : (blockIdx.z == 2) ? d2 : d3;
  int x  = blockIdx.x * 32 + threadIdx.x;
  int y0 = blockIdx.y * 32 + threadIdx.y;
  for (int j = 0; j < 32; j += 8)
    tile[threadIdx.y + j][threadIdx.x] = src[(size_t)(y0 + j) * DD + x];
  __syncthreads();
  int x2 = blockIdx.y * 32 + threadIdx.x;
  int y2 = blockIdx.x * 32 + threadIdx.y;
  for (int j = 0; j < 32; j += 8)
    dst[(size_t)(y2 + j) * DD + x2] = f2bf(tile[threadIdx.x][threadIdx.y + j]);
}

// ---------------- tiled bf16 GEMM with async double-buffered LDS ------------
// Y(m,n) = sum_k A(m,k)*Bt(n,k)
// mode 0: bf16 Yb[m*DD+n]
// mode 1: bf16 Vt[((b*H+h)*64+d)*S + s]   (transposed-per-head V)
// mode 2: f32  Yf[m*DD+n] = acc + bias[n]
#define ATILE (128 * 72)
#define BTILE ( 64 * 72)
__global__ __launch_bounds__(256)
void gemm_bf16(const unsigned short* __restrict__ A,
               const unsigned short* __restrict__ Bt,
               unsigned short* __restrict__ Yb, float* __restrict__ Yf,
               const float* __restrict__ bias, int mode) {
  __shared__ unsigned short At [2 * ATILE];
  __shared__ unsigned short Btl[2 * BTILE];
  int tid = threadIdx.x;
  int lane = tid & 31, w = tid >> 5;
  int wm = (w >> 1) * 32, wn = (w & 1) * 32;
  int hf = lane >> 4, ln = lane & 15;
  size_t mTile = (size_t)blockIdx.x * 128;
  size_t nTile = (size_t)blockIdx.y * 64;

  int ldRow = tid >> 3, ldCh = (tid & 7) * 8;
  unsigned aLds0 = (unsigned)(uintptr_t)(At);
  unsigned bLds0 = (unsigned)(uintptr_t)(Btl);

  // issue one 64-wide K-slab into LDS buffer `buf` (async, no VGPR staging)
  auto issueTile = [&](int k0, int buf) {
    unsigned aB = aLds0 + (unsigned)(buf * ATILE) * 2u;
    unsigned bB = bLds0 + (unsigned)(buf * BTILE) * 2u;
    for (int r = 0; r < 128; r += 32)
      async_b128(aB + (unsigned)((ldRow + r) * 72 + ldCh) * 2u,
                 A + (mTile + ldRow + r) * DD + k0 + ldCh);
    for (int r = 0; r < 64; r += 32)
      async_b128(bB + (unsigned)((ldRow + r) * 72 + ldCh) * 2u,
                 Bt + (nTile + ldRow + r) * DD + k0 + ldCh);
  };

  v8f acc[2][2];
  for (int a = 0; a < 2; a++)
    for (int c = 0; c < 2; c++)
      for (int j = 0; j < 8; j++) acc[a][c][j] = 0.f;

  issueTile(0, 0);
  wait_async0();
  __syncthreads();

  for (int k0 = 0; k0 < DD; k0 += 64) {
    int buf = (k0 >> 6) & 1;
    if (k0 + 64 < DD) issueTile(k0 + 64, buf ^ 1);

    const unsigned short* Ab = At  + buf * ATILE;
    const unsigned short* Bb = Btl + buf * BTILE;
    int klo = hf * 8, khi = hf * 8;
    for (int ks = 0; ks < 64; ks += 32) {
      Frag16 a0, a1, b0, b1;
      a0.u[0] = *(const uint4*)(Ab + (wm + ln) * 72 + ks + klo);
      a0.u[1] = *(const uint4*)(Ab + (wm + ln) * 72 + ks + klo + 16);
      a1.u[0] = *(const uint4*)(Ab + (wm + 16 + ln) * 72 + ks + klo);
      a1.u[1] = *(const uint4*)(Ab + (wm + 16 + ln) * 72 + ks + klo + 16);
      b0.u[0] = *(const uint4*)(Bb + (wn + ln) * 72 + ks + khi);
      b0.u[1] = *(const uint4*)(Bb + (wn + ln) * 72 + ks + khi + 16);
      b1.u[0] = *(const uint4*)(Bb + (wn + 16 + ln) * 72 + ks + khi);
      b1.u[1] = *(const uint4*)(Bb + (wn + 16 + ln) * 72 + ks + khi + 16);
      acc[0][0] = __builtin_amdgcn_wmma_f32_16x16x32_bf16(false, a0.v, false, b0.v, (short)0, acc[0][0], false, false);
      acc[0][1] = __builtin_amdgcn_wmma_f32_16x16x32_bf16(false, a0.v, false, b1.v, (short)0, acc[0][1], false, false);
      acc[1][0] = __builtin_amdgcn_wmma_f32_16x16x32_bf16(false, a1.v, false, b0.v, (short)0, acc[1][0], false, false);
      acc[1][1] = __builtin_amdgcn_wmma_f32_16x16x32_bf16(false, a1.v, false, b1.v, (short)0, acc[1][1], false, false);
    }

    if (k0 + 64 < DD) wait_async0();  // own next-slab transfers done
    __syncthreads();                  // everyone's transfers + reads done
  }

  for (int ms = 0; ms < 2; ms++) {
    for (int ns = 0; ns < 2; ns++) {
      size_t mBase = mTile + wm + ms * 16 + hf * 8;
      size_t n = nTile + wn + ns * 16 + ln;
      v8f c = acc[ms][ns];
      if (mode == 0) {
        for (int j = 0; j < 8; j++) Yb[(mBase + j) * DD + n] = f2bf(c[j]);
      } else if (mode == 1) {
        int bb = (int)(mBase / SS), s0i = (int)(mBase % SS);
        int h = (int)(n >> 6), d = (int)(n & 63);
        union { unsigned short s[8]; uint4 u; } pk;
        for (int j = 0; j < 8; j++) pk.s[j] = f2bf(c[j]);
        *(uint4*)(Yb + (((size_t)(bb * HH + h) * HDm + d) * SS + s0i)) = pk.u;
      } else {
        float bv = bias[n];
        for (int j = 0; j < 8; j++) Yf[(mBase + j) * DD + n] = c[j] + bv;
      }
    }
  }
}

// ---------------- flash attention, 1 wave = 16 queries, all-WMMA ------------
__global__ __launch_bounds__(128)
void attn_kernel(const unsigned short* __restrict__ Q,
                 const unsigned short* __restrict__ K,
                 const unsigned short* __restrict__ Vt,
                 unsigned short* __restrict__ Ctx) {
  int tid = threadIdx.x;
  int lane = tid & 31, w = tid >> 5;
  int hf = lane >> 4, ln = lane & 15;
  int qt = blockIdx.x * 4 + w;
  int h = blockIdx.y, b = blockIdx.z;
  int qBase = qt * 16;
  size_t bS = (size_t)b * SS;

  // Q as B-fragments (lane = query column, table-order d)
  Frag16 bq0, bq1;
  {
    const unsigned short* qp = Q + (bS + qBase + ln) * DD + h * HDm + hf * 8;
    bq0.u[0] = *(const uint4*)(qp);
    bq0.u[1] = *(const uint4*)(qp + 16);
    bq1.u[0] = *(const uint4*)(qp + 32);
    bq1.u[1] = *(const uint4*)(qp + 48);
  }

  v8f accO[4];
  for (int d = 0; d < 4; d++)
    for (int j = 0; j < 8; j++) accO[d][j] = 0.f;
  float m_run = -1e30f, l_run = 0.f;
  const float cs = 0.125f * 1.44269504f;  // (1/sqrt(64)) * log2(e)
  int q_glob = qBase + ln;
  int nblk = (qBase + 16 + 31) >> 5;

  const unsigned short* Kbase = K + bS * DD + h * HDm;
  const unsigned short* Vbase = Vt + ((size_t)(b * HH + h) * HDm + ln) * SS;

  for (int kb = 0; kb < nblk; kb++) {
    int kBase = kb * 32;
    // prefetch next key-block lines while this block computes
    if (kb + 1 < nblk) {
      __builtin_prefetch(Kbase + (size_t)(kBase + 32 + ln) * DD, 0, 3);
      __builtin_prefetch(Vbase + kBase + 32, 0, 3);
    }
    v8f s0, s1;
    for (int j = 0; j < 8; j++) { s0[j] = 0.f; s1[j] = 0.f; }
    {  // S^T(key,q) = K(key,:) . Q(q,:)
      Frag16 ak;
      const unsigned short* kp0 = Kbase + (size_t)(kBase + ln) * DD + hf * 8;
      const unsigned short* kp1 = kp0 + (size_t)16 * DD;
      ak.u[0] = *(const uint4*)(kp0);      ak.u[1] = *(const uint4*)(kp0 + 16);
      s0 = __builtin_amdgcn_wmma_f32_16x16x32_bf16(false, ak.v, false, bq0.v, (short)0, s0, false, false);
      ak.u[0] = *(const uint4*)(kp0 + 32); ak.u[1] = *(const uint4*)(kp0 + 48);
      s0 = __builtin_amdgcn_wmma_f32_16x16x32_bf16(false, ak.v, false, bq1.v, (short)0, s0, false, false);
      ak.u[0] = *(const uint4*)(kp1);      ak.u[1] = *(const uint4*)(kp1 + 16);
      s1 = __builtin_amdgcn_wmma_f32_16x16x32_bf16(false, ak.v, false, bq0.v, (short)0, s1, false, false);
      ak.u[0] = *(const uint4*)(kp1 + 32); ak.u[1] = *(const uint4*)(kp1 + 48);
      s1 = __builtin_amdgcn_wmma_f32_16x16x32_bf16(false, ak.v, false, bq1.v, (short)0, s1, false, false);
    }
    // causal mask: key row index = kBase + j + 8*hf (+16 for s1)
    for (int j = 0; j < 8; j++) {
      int km = kBase + j + hf * 8;
      if (km > q_glob)      s0[j] = -1e30f;
      if (km + 16 > q_glob) s1[j] = -1e30f;
    }
    // online softmax (per-lane over 16 keys, +shfl_xor16 for the other half)
    float bm = s0[0];
    for (int j = 1; j < 8; j++) bm = fmaxf(bm, s0[j]);
    for (int j = 0; j < 8; j++) bm = fmaxf(bm, s1[j]);
    bm = fmaxf(bm, __shfl_xor(bm, 16, 32));
    float m_new = fmaxf(m_run, bm);
    float alpha = exp2f((m_run - m_new) * cs);
    float p0[8], p1[8], rs = 0.f;
    for (int j = 0; j < 8; j++) {
      p0[j] = exp2f((s0[j] - m_new) * cs);
      p1[j] = exp2f((s1[j] - m_new) * cs);
      rs += p0[j] + p1[j];
    }
    rs += __shfl_xor(rs, 16, 32);
    l_run = l_run * alpha + rs;
    m_run = m_new;
    for (int d = 0; d < 4; d++)
      for (int j = 0; j < 8; j++) accO[d][j] *= alpha;
    // build P^T B-fragment in registers (lane=query, 16 contiguous keys)
    Frag16 pf;
    for (int e = 0; e < 8; e++) {
      float o0 = __shfl_xor(p0[e], 16, 32);
      float o1 = __shfl_xor(p1[e], 16, 32);
      pf.s[e]     = f2bf(hf ? o1 : p0[e]);
      pf.s[e + 8] = f2bf(hf ? p1[e] : o0);
    }
    // O^T(d,q) += Vt(d,key) * P^T(key,q)
    const unsigned short* vp = Vbase + kBase + hf * 16;
    for (int d = 0; d < 4; d++) {
      Frag16 av;
      const unsigned short* vpd = vp + (size_t)(d * 16) * SS;
      av.u[0] = *(const uint4*)(vpd);
      av.u[1] = *(const uint4*)(vpd + 8);
      accO[d] = __builtin_amdgcn_wmma_f32_16x16x32_bf16(false, av.v, false, pf.v, (short)0, accO[d], false, false);
    }
  }

  float rl = 1.0f / l_run;
  unsigned short* cp = Ctx + (bS + qBase + ln) * DD + h * HDm + hf * 8;
  for (int d = 0; d < 4; d++) {
    union { unsigned short s[8]; uint4 u; } pk;
    for (int j = 0; j < 8; j++) pk.s[j] = f2bf(accO[d][j] * rl);
    *(uint4*)(cp + d * 16) = pk.u;
  }
}

extern "C" void kernel_launch(void* const* d_in, const int* in_sizes, int n_in,
                              void* d_out, int out_size, void* d_ws, size_t ws_size,
                              hipStream_t stream) {
  (void)in_sizes; (void)n_in; (void)out_size; (void)ws_size;
  const float* x  = (const float*)d_in[0];
  const float* Wq = (const float*)d_in[1];
  const float* Wk = (const float*)d_in[2];
  const float* Wv = (const float*)d_in[3];
  const float* Wo = (const float*)d_in[4];
  const float* bo = (const float*)d_in[5];
  float* out = (float*)d_out;

  char* ws = (char*)d_ws;
  size_t off = 0;
  auto carve = [&](size_t bytes) {
    void* p = ws + off;
    off += (bytes + 255) & ~(size_t)255;
    return p;
  };
  unsigned short* xbf = (unsigned short*)carve((size_t)MM * DD * 2);
  unsigned short* WqT = (unsigned short*)carve((size_t)DD * DD * 2);
  unsigned short* WkT = (unsigned short*)carve((size_t)DD * DD * 2);
  unsigned short* WvT = (unsigned short*)carve((size_t)DD * DD * 2);
  unsigned short* WoT = (unsigned short*)carve((size_t)DD * DD * 2);
  unsigned short* Qbf = (unsigned short*)carve((size_t)MM * DD * 2);
  unsigned short* Kbf = (unsigned short*)carve((size_t)MM * DD * 2);
  unsigned short* Vt  = (unsigned short*)carve((size_t)MM * DD * 2);
  unsigned short* ctx = (unsigned short*)carve((size_t)MM * DD * 2);

  int n4 = MM * DD / 4;
  cvt_f32_bf16<<<(n4 + 255) / 256, 256, 0, stream>>>(x, xbf, n4);
  transpose_w_bf16<<<dim3(DD / 32, DD / 32, 4), dim3(32, 8), 0, stream>>>(
      Wq, Wk, Wv, Wo, WqT, WkT, WvT, WoT);

  dim3 gg(MM / 128, DD / 64);
  gemm_bf16<<<gg, 256, 0, stream>>>(xbf, WqT, Qbf, nullptr, nullptr, 0);
  gemm_bf16<<<gg, 256, 0, stream>>>(xbf, WkT, Kbf, nullptr, nullptr, 0);
  gemm_bf16<<<gg, 256, 0, stream>>>(xbf, WvT, Vt,  nullptr, nullptr, 1);

  attn_kernel<<<dim3(SS / 64, HH, BB), 128, 0, stream>>>(Qbf, Kbf, Vt, ctx);

  gemm_bf16<<<gg, 256, 0, stream>>>(ctx, WoT, nullptr, out, bo, 2);
}
